// DKVMN_33809982554170
// MI455X (gfx1250) — matrix operand order repaired and checked by reading, not compiled, not run on hardware
//
#include <hip/hip_runtime.h>
#include <hip/hip_bf16.h>

// ---------------- problem constants ----------------
constexpr int S    = 512;
constexpr int B    = 256;
constexpr int MEM  = 50;
constexpr int HID  = 256;
constexpr int QD   = 128;
constexpr int SD   = 64;
constexpr int KEY  = QD + SD;        // 192
constexpr int QAD  = QD + 4 * SD;    // 384
constexpr int PRED = HID + 2*SD + QD;// 512
constexpr int NSUB = 8;
constexpr int N    = S * B;          // 131072 rows

// ---------------- types ----------------
typedef unsigned int       u32;
typedef unsigned short     u16;
typedef __attribute__((ext_vector_type(16))) __bf16 v16bf;
typedef __attribute__((ext_vector_type(8)))  float  v8f;
typedef __attribute__((ext_vector_type(4)))  u32    u32x4;

struct Frag32 { u32x4 lo; u32x4 hi; };   // 32 bytes == v16bf

// bf16 helpers (RNE)
__device__ __forceinline__ u16 f2bf(float x) {
    u32 u = __builtin_bit_cast(u32, x);
    u32 r = u + 0x7FFFu + ((u >> 16) & 1u);
    return (u16)(r >> 16);
}
__device__ __forceinline__ float bf2f(u16 h) {
    return __builtin_bit_cast(float, (u32)h << 16);
}

// ---------------- WMMA fragment loaders ----------------
// A (16x32 bf16, ISA 7.12.2): lane L (r=L&15, base = L>=16 ? 8 : 0) holds
// row m0+r, K = k0+base+{0..7} (elems 0..7) and k0+16+base+{0..7} (elems 8..15).
__device__ __forceinline__ v16bf load_a_frag(const u16* X, int ldx, int m0, int k0, int lane) {
    int r    = lane & 15;
    int base = (lane >> 4) ? 8 : 0;
    const u16* p = X + (size_t)(m0 + r) * ldx + k0 + base;
    Frag32 fr;
    fr.lo = *(const u32x4*)(p);
    fr.hi = *(const u32x4*)(p + 16);
    return __builtin_bit_cast(v16bf, fr);
}
// B fragments are pre-packed: fragment f, lane L owns 16 contiguous bf16.
__device__ __forceinline__ v16bf load_b_frag(const u16* BP, int f, int lane) {
    const u16* p = BP + ((size_t)f * 32 + lane) * 16;
    Frag32 fr;
    fr.lo = *(const u32x4*)(p);
    fr.hi = *(const u32x4*)(p + 8);
    return __builtin_bit_cast(v16bf, fr);
}
__device__ __forceinline__ v8f wmma_bf16(v16bf a, v16bf b, v8f c) {
    return __builtin_amdgcn_wmma_f32_16x16x32_bf16(false, a, false, b, (short)0, c, false, false);
}

// ---------------- weight pack: f32 (K,H) row-major -> B-fragment bf16 ----------------
// B (32x16 bf16) layout (C-analog): lane L holds N = nt*16+(L&15),
// elem e -> K = kc*32 + (e&7) + (L>=16 ? 8 : 0) + (e>=8 ? 16 : 0).
__global__ void pack_b_kernel(const float* __restrict__ W, u16* __restrict__ BP, int K, int H) {
    int idx = blockIdx.x * blockDim.x + threadIdx.x;
    if (idx >= K * H) return;
    int e    = idx & 15;
    int lane = (idx >> 4) & 31;
    int f    = idx >> 9;
    int KC   = K >> 5;
    int kc   = f % KC;
    int nt   = f / KC;
    int n = nt * 16 + (lane & 15);
    int k = kc * 32 + (e & 7) + ((lane >= 16) ? 8 : 0) + ((e >= 8) ? 16 : 0);
    BP[((size_t)f * 32 + lane) * 16 + e] = f2bf(W[(size_t)k * H + n]);
}

// ---------------- K0: embeddings + concat (writes QA and static cols of PRED_IN) ----------------
__global__ void embed_kernel(const int* __restrict__ ans, const int* __restrict__ cans,
                             const int* __restrict__ labels, const float* __restrict__ mask,
                             const int* __restrict__ qids, const int* __restrict__ sids,
                             const float* __restrict__ smask,
                             const float* __restrict__ q_emb, const float* __restrict__ s_emb,
                             const float* __restrict__ a_emb, const float* __restrict__ l_emb,
                             u16* __restrict__ QA, u16* __restrict__ P) {
    int t   = threadIdx.x & 63;
    int row = blockIdx.x * 4 + (threadIdx.x >> 6);
    float m = mask[row];
    int a = ans[row], c = cans[row], l = labels[row];
    float subj = 0.f;
    #pragma unroll
    for (int j = 0; j < NSUB; ++j) {
        int sid = sids[(size_t)row * NSUB + j];
        subj += s_emb[(size_t)sid * SD + t] * smask[(size_t)row * NSUB + j];
    }
    size_t qb = (size_t)qids[row] * QD;
    float q0 = q_emb[qb + t];
    float q1 = q_emb[qb + 64 + t];
    float ae = a_emb[a * SD + t] * m;
    float ce = a_emb[c * SD + t];
    float le = l_emb[l * SD + t] * m;
    u16* qa = QA + (size_t)row * QAD;   // [ans_e | ca_e | lab_e | q | subj]
    qa[t]        = f2bf(ae);
    qa[64 + t]   = f2bf(ce);
    qa[128 + t]  = f2bf(le);
    qa[192 + t]  = f2bf(q0);
    qa[256 + t]  = f2bf(q1);
    qa[320 + t]  = f2bf(subj);
    u16* p = P + (size_t)row * PRED;    // [q | subj | (read, from scan) | ca_e]
    p[t]       = f2bf(q0);
    p[64 + t]  = f2bf(q1);
    p[128 + t] = f2bf(subj);
    p[448 + t] = f2bf(ce);
}

// ---------------- K1: attention weights w = softmax(qs @ mem_key^T) ----------------
__global__ void attn_w_kernel(const u16* __restrict__ QA, const float* __restrict__ mem_key,
                              float* __restrict__ W) {
    __shared__ float qs[KEY];
    __shared__ float ex[64];
    __shared__ float red;
    int row = blockIdx.x;
    int t   = threadIdx.x;
    const u16* q = QA + (size_t)row * QAD + 192;   // qs is the tail of qa
    for (int i = t; i < KEY; i += 64) qs[i] = bf2f(q[i]);
    __syncthreads();
    float d = 0.f;
    if (t < MEM) {
        const float* mk = mem_key + (size_t)t * KEY;
        #pragma unroll 4
        for (int k = 0; k < KEY; ++k) d += qs[k] * mk[k];
    }
    ex[t] = (t < MEM) ? d : -1e30f;
    __syncthreads();
    if (t == 0) {
        float mx = -1e30f;
        for (int i = 0; i < MEM; ++i) mx = fmaxf(mx, ex[i]);
        red = mx;
    }
    __syncthreads();
    float e = (t < MEM) ? __expf(d - red) : 0.f;
    ex[t] = e;
    __syncthreads();
    if (t == 0) {
        float sm = 0.f;
        for (int i = 0; i < MEM; ++i) sm += ex[i];
        red = 1.f / sm;
    }
    __syncthreads();
    if (t < MEM) W[(size_t)row * MEM + t] = e * red;
}

// ---------------- K2: dual GEMM  E=sigmoid(QA@We+be), A=tanh(QA@Wa+ba)  (WMMA) ----------------
// Wave tile: M=32, N=32 for BOTH outputs. 8 waves tile 128(M) x 64(N) per block.
// Per K-chunk: 2 A-frags + 2 We-frags + 2 Wa-frags feed 8 WMMAs.
__global__ void gemm_ea_kernel(const u16* __restrict__ QA,
                               const u16* __restrict__ WeP, const u16* __restrict__ WaP,
                               const float* __restrict__ be, const float* __restrict__ ba,
                               u16* __restrict__ E, u16* __restrict__ A) {
    constexpr int K = QAD, H = HID, KC = K / 32;
    int lane = threadIdx.x & 31;
    int wv   = threadIdx.x >> 5;
    int wv_m = wv & 3, wv_n = wv >> 2;
    int m0   = blockIdx.x * 128 + wv_m * 32;
    int n0   = blockIdx.y * 64  + wv_n * 32;
    int fb0  = ((n0 >> 4) + 0) * KC;
    int fb1  = ((n0 >> 4) + 1) * KC;
    v8f aE[2][2] = {}; v8f aA[2][2] = {};
    for (int kc = 0; kc < KC; ++kc) {
        if (kc + 1 < KC)   // one row per lane covers the whole 32-row M tile
            __builtin_prefetch(QA + (size_t)(m0 + lane) * K + (kc + 1) * 32, 0, 3);
        v16bf a0  = load_a_frag(QA, K, m0,      kc * 32, lane);
        v16bf a1  = load_a_frag(QA, K, m0 + 16, kc * 32, lane);
        v16bf be0 = load_b_frag(WeP, fb0 + kc, lane);
        v16bf be1 = load_b_frag(WeP, fb1 + kc, lane);
        v16bf ba0 = load_b_frag(WaP, fb0 + kc, lane);
        v16bf ba1 = load_b_frag(WaP, fb1 + kc, lane);
        aE[0][0] = wmma_bf16(a0, be0, aE[0][0]);
        aE[0][1] = wmma_bf16(a0, be1, aE[0][1]);
        aE[1][0] = wmma_bf16(a1, be0, aE[1][0]);
        aE[1][1] = wmma_bf16(a1, be1, aE[1][1]);
        aA[0][0] = wmma_bf16(a0, ba0, aA[0][0]);
        aA[0][1] = wmma_bf16(a0, ba1, aA[0][1]);
        aA[1][0] = wmma_bf16(a1, ba0, aA[1][0]);
        aA[1][1] = wmma_bf16(a1, ba1, aA[1][1]);
    }
    #pragma unroll
    for (int i = 0; i < 2; ++i) {
        int rbase = m0 + i * 16 + ((lane >= 16) ? 8 : 0);
        #pragma unroll
        for (int j = 0; j < 2; ++j) {
            int col = n0 + j * 16 + (lane & 15);
            float bE = be[col], bA = ba[col];
            #pragma unroll
            for (int r = 0; r < 8; ++r) {
                size_t o = (size_t)(rbase + r) * H + col;
                float xe = aE[i][j][r] + bE;
                float xa = aA[i][j][r] + bA;
                E[o] = f2bf(1.f / (1.f + __expf(-xe)));
                A[o] = f2bf(tanhf(xa));
            }
        }
    }
}

// ---------------- K3: recurrent scan, M resident in VGPRs, one block per batch b ----------------
__global__ void scan_kernel(const float* __restrict__ W, const u16* __restrict__ E,
                            const u16* __restrict__ A, const float* __restrict__ mem_val0,
                            u16* __restrict__ P) {
    int b = blockIdx.x;    // 0..B-1
    int h = threadIdx.x;   // 0..HID-1
    float M[MEM];
    #pragma unroll
    for (int m = 0; m < MEM; ++m) M[m] = mem_val0[(size_t)m * HID + h];
    __shared__ float ws[2][MEM];
    if (h < MEM) ws[0][h] = W[((size_t)0 * B + b) * MEM + h];
    __syncthreads();
    for (int s = 0; s < S; ++s) {
        size_t row = (size_t)s * B + b;
        if (s + 1 < S && h < MEM)                       // prefetch next step's w
            ws[(s + 1) & 1][h] = W[((size_t)(s + 1) * B + b) * MEM + h];
        float e = bf2f(E[row * HID + h]);
        float a = bf2f(A[row * HID + h]);
        const float* w = ws[s & 1];
        float read = 0.f;
        #pragma unroll
        for (int m = 0; m < MEM; ++m) {
            float wm = w[m];
            read = fmaf(wm, M[m], read);
            M[m] = M[m] * (1.f - wm * e) + wm * a;
        }
        P[row * PRED + 192 + h] = f2bf(read);           // forward_ht columns of pred_input
        __syncthreads();
    }
}

// ---------------- K4a/b: GEMM + bias + relu (WMMA), X(N,K) @ Wp -> Y(N,H) bf16 ----------------
// Wave tile: M=32, N=64. 8 waves tile 128(M) x 128(N) per block.
// Per K-chunk: 2 A-frags + 4 B-frags feed 8 WMMAs.
__global__ void gemm_bias_relu_kernel(const u16* __restrict__ X, const u16* __restrict__ BP,
                                      const float* __restrict__ bias, u16* __restrict__ Y,
                                      int K, int H) {
    int KC   = K >> 5;
    int lane = threadIdx.x & 31;
    int wv   = threadIdx.x >> 5;
    int wv_m = wv & 3, wv_n = wv >> 2;
    int m0   = blockIdx.x * 128 + wv_m * 32;
    int n0   = blockIdx.y * 128 + wv_n * 64;
    int fb[4];
    #pragma unroll
    for (int j = 0; j < 4; ++j) fb[j] = ((n0 >> 4) + j) * KC;
    v8f acc[2][4] = {};
    for (int kc = 0; kc < KC; ++kc) {
        if (kc + 1 < KC)   // one row per lane covers the whole 32-row M tile
            __builtin_prefetch(X + (size_t)(m0 + lane) * K + (kc + 1) * 32, 0, 3);
        v16bf a0 = load_a_frag(X, K, m0,      kc * 32, lane);
        v16bf a1 = load_a_frag(X, K, m0 + 16, kc * 32, lane);
        #pragma unroll
        for (int j = 0; j < 4; ++j) {
            v16bf b = load_b_frag(BP, fb[j] + kc, lane);
            acc[0][j] = wmma_bf16(a0, b, acc[0][j]);
            acc[1][j] = wmma_bf16(a1, b, acc[1][j]);
        }
    }
    #pragma unroll
    for (int i = 0; i < 2; ++i) {
        int rbase = m0 + i * 16 + ((lane >= 16) ? 8 : 0);
        #pragma unroll
        for (int j = 0; j < 4; ++j) {
            int col = n0 + j * 16 + (lane & 15);
            float bs = bias[col];
            #pragma unroll
            for (int r = 0; r < 8; ++r) {
                float v = acc[i][j][r] + bs;
                Y[(size_t)(rbase + r) * H + col] = f2bf(fmaxf(v, 0.f));
            }
        }
    }
}

// ---------------- K4c: logits = ((H2 + P) @ Wo + bo), one wave per row ----------------
__global__ void logits_kernel(const u16* __restrict__ H2, const u16* __restrict__ P,
                              const float* __restrict__ Wo, const float* __restrict__ bo,
                              float* __restrict__ out) {
    int lane = threadIdx.x & 31;
    int row  = blockIdx.x * 4 + (threadIdx.x >> 5);
    const u16* h = H2 + (size_t)row * PRED;
    const u16* p = P  + (size_t)row * PRED;
    float acc = 0.f;
    #pragma unroll 4
    for (int k = lane; k < PRED; k += 32)
        acc = fmaf(bf2f(h[k]) + bf2f(p[k]), Wo[k], acc);
    #pragma unroll
    for (int off = 16; off > 0; off >>= 1)
        acc += __shfl_xor(acc, off, 32);
    if (lane == 0) out[row] = acc + bo[0];
}

// ---------------- host side ----------------
extern "C" void kernel_launch(void* const* d_in, const int* in_sizes, int n_in,
                              void* d_out, int out_size, void* d_ws, size_t ws_size,
                              hipStream_t stream) {
    const int*   ans      = (const int*)  d_in[0];
    const int*   cans     = (const int*)  d_in[1];
    const int*   labels   = (const int*)  d_in[2];
    const float* mask     = (const float*)d_in[3];
    const int*   qids     = (const int*)  d_in[4];
    const int*   sids     = (const int*)  d_in[5];
    const float* smask    = (const float*)d_in[6];
    const float* q_emb    = (const float*)d_in[7];
    const float* s_emb    = (const float*)d_in[8];
    const float* a_emb    = (const float*)d_in[9];
    const float* l_emb    = (const float*)d_in[10];
    const float* mem_key  = (const float*)d_in[11];
    const float* mem_val0 = (const float*)d_in[12];
    const float* We       = (const float*)d_in[13];
    const float* be       = (const float*)d_in[14];
    const float* Wa       = (const float*)d_in[15];
    const float* ba       = (const float*)d_in[16];
    const float* W1       = (const float*)d_in[17];
    const float* b1       = (const float*)d_in[18];
    const float* W2       = (const float*)d_in[19];
    const float* b2       = (const float*)d_in[20];
    const float* Wo       = (const float*)d_in[21];
    const float* bo       = (const float*)d_in[22];
    float* out = (float*)d_out;

    char* ws = (char*)d_ws;
    size_t off = 0;
    auto carve = [&](size_t bytes) -> void* {
        off = (off + 255) & ~(size_t)255;
        void* p = ws + off;
        off += bytes;
        return p;
    };
    u16* QA  = (u16*)carve((size_t)N * QAD * 2);
    u16* Eb  = (u16*)carve((size_t)N * HID * 2);   // E and A contiguous:
    u16* Ab  = (u16*)carve((size_t)N * HID * 2);   //   region reused as H1 after scan
    u16* P   = (u16*)carve((size_t)N * PRED * 2);
    float* Wt= (float*)carve((size_t)N * MEM * 4);
    u16* H2  = (u16*)carve((size_t)N * PRED * 2);
    u16* WeP = (u16*)carve((size_t)QAD * HID * 2);
    u16* WaP = (u16*)carve((size_t)QAD * HID * 2);
    u16* W1P = (u16*)carve((size_t)PRED * PRED * 2);
    u16* W2P = (u16*)carve((size_t)PRED * PRED * 2);
    u16* H1  = Eb;   // alias: E|A (2*N*HID*2 bytes) == N*PRED*2 bytes

    // weight packing (tiny)
    pack_b_kernel<<<(QAD * HID + 255) / 256, 256, 0, stream>>>(We, WeP, QAD, HID);
    pack_b_kernel<<<(QAD * HID + 255) / 256, 256, 0, stream>>>(Wa, WaP, QAD, HID);
    pack_b_kernel<<<(PRED * PRED + 255) / 256, 256, 0, stream>>>(W1, W1P, PRED, PRED);
    pack_b_kernel<<<(PRED * PRED + 255) / 256, 256, 0, stream>>>(W2, W2P, PRED, PRED);

    // embeddings + concat
    embed_kernel<<<N / 4, 256, 0, stream>>>(ans, cans, labels, mask, qids, sids, smask,
                                            q_emb, s_emb, a_emb, l_emb, QA, P);
    // attention softmax weights
    attn_w_kernel<<<N, 64, 0, stream>>>(QA, mem_key, Wt);
    // hoisted erase/add GEMMs (WMMA), block tile 128x64, wave tile 32x32 (dual output)
    gemm_ea_kernel<<<dim3(N / 128, HID / 64), 256, 0, stream>>>(QA, WeP, WaP, be, ba, Eb, Ab);
    // recurrent scan: 1 block per batch element, M in VGPRs
    scan_kernel<<<B, HID, 0, stream>>>(Wt, Eb, Ab, mem_val0, P);
    // prediction MLP (WMMA), block tile 128x128, wave tile 32x64
    gemm_bias_relu_kernel<<<dim3(N / 128, PRED / 128), 256, 0, stream>>>(P,  W1P, b1, H1, PRED, PRED);
    gemm_bias_relu_kernel<<<dim3(N / 128, PRED / 128), 256, 0, stream>>>(H1, W2P, b2, H2, PRED, PRED);
    // residual + output head
    logits_kernel<<<N / 4, 128, 0, stream>>>(H2, P, Wo, bo, out);
}
